// MultilayerGINRegression_25177098289491
// MI455X (gfx1250) — matrix operand order repaired and checked by reading, not compiled, not run on hardware
//
#include <hip/hip_runtime.h>

typedef __attribute__((ext_vector_type(16))) __bf16 v16bf;
typedef __attribute__((ext_vector_type(8)))  float  v8f;
typedef __attribute__((ext_vector_type(4)))  unsigned int v4u;
typedef __attribute__((ext_vector_type(8)))  int  v8i;
typedef __attribute__((ext_vector_type(4)))  int  v4i;

#define N_NODES  50000
#define N_EDGES  1000000
#define N_GRAPHS 64
#define IN_FEAT  128
#define STRIDE   512   // fixed row stride (floats) for all node buffers

#if __has_builtin(__builtin_amdgcn_tensor_load_to_lds) && \
    __has_builtin(__builtin_amdgcn_s_wait_tensorcnt)
#define USE_TDM 1
#else
#define USE_TDM 0
#endif

// ---------------- wave32 helpers ----------------
__device__ __forceinline__ float wredsum(float v) {
  #pragma unroll
  for (int off = 16; off > 0; off >>= 1) v += __shfl_xor(v, off, 32);
  return v;
}

// exp_map_zero on spatial vector: radial scale factor (includes hnormalize MAX_NORM clip)
__device__ __forceinline__ float exp_scale(float s2, float c) {
  float nrm = sqrtf(fmaxf(s2 + 1e-7f, 1e-6f));
  float cut = fminf(nrm, 50.0f);
  float sc  = sqrtf(c);
  float t   = sc * sinhf(cut / sc) / nrm;          // tail scale
  float tn  = t * sqrtf(s2);                       // tail norm
  float f   = fminf(1.0f, 1000.0f / fmaxf(tn, 1e-8f));
  return t * f;
}

// log_map_zero on hyperboloid point with spatial norm^2 s2 (x0 = sqrt(c+s2)): radial scale
__device__ __forceinline__ float log_scale(float s2, float c) {
  float sc = sqrtf(c);
  float y0 = sqrtf(c + s2);
  float d  = sc * acoshf(fmaxf(y0 / sc + 1e-7f, 1.0f + 1e-7f));
  float n  = sqrtf(s2 + 1e-7f);
  return d / n;
}

// ---------------- fused radial-chain kernel (wave per node) ----------------
// one stage : out = log_cB(exp_cA(in))
// two stages: out = log_cD(exp_cC(relu(log_cB(exp_cA(in)))))
__global__ __launch_bounds__(256)
void k_chain(const float* __restrict__ in, float* __restrict__ out,
             float* __restrict__ aggcopy, int in_stride, int dpad,
             float cA, float cB, int two_stage, float cC, float cD) {
  int node = blockIdx.x * 8 + (threadIdx.x >> 5);
  if (node >= N_NODES) return;
  int lane = threadIdx.x & 31;
  const float* ip = in + (size_t)node * in_stride;
  int cnt = dpad >> 5;
  float v[16];
  float s2 = 0.f;
  #pragma unroll
  for (int i = 0; i < 16; ++i) {
    v[i] = 0.f;
    if (i < cnt) { float x = ip[lane + 32 * i]; v[i] = x; s2 += x * x; }
  }
  s2 = wredsum(s2);
  float s = exp_scale(s2, cA);
  s *= log_scale(s2 * s * s, cB);          // radial: norm scales exactly by s
  if (two_stage) {
    float n2 = 0.f;
    #pragma unroll
    for (int i = 0; i < 16; ++i) { v[i] = fmaxf(v[i] * s, 0.f); n2 += v[i] * v[i]; }
    n2 = wredsum(n2);
    float s3 = exp_scale(n2, cC);
    s = s3 * log_scale(n2 * s3 * s3, cD);
  }
  float* op = out + (size_t)node * STRIDE;
  float* ap = aggcopy ? aggcopy + (size_t)node * STRIDE : nullptr;
  #pragma unroll
  for (int i = 0; i < 16; ++i) {
    if (i < cnt) {
      float r = v[i] * s;
      op[lane + 32 * i] = r;
      if (ap) ap[lane + 32 * i] = r;       // segment_sum init = self term (GIN eps = 0)
    }
  }
}

// ---------------- edge scatter: agg[dst] += vt[src] ----------------
__global__ __launch_bounds__(256)
void k_scatter(const float* __restrict__ vt, float* __restrict__ agg,
               const int* __restrict__ src, const int* __restrict__ dst, int lc) {
  int idx = blockIdx.x * 256 + threadIdx.x;
  int total = N_EDGES << lc;                 // lc = log2(dpad/4)
  if (idx >= total) return;
  int e  = idx >> lc;
  int ch = idx & ((1 << lc) - 1);
  int s = src[e], d = dst[e];
  const float4 val = *(const float4*)(vt + (size_t)s * STRIDE + ch * 4);
  float* ap = agg + (size_t)d * STRIDE + ch * 4;
  atomicAdd(ap + 0, val.x);
  atomicAdd(ap + 1, val.y);
  atomicAdd(ap + 2, val.z);
  atomicAdd(ap + 3, val.w);
}

// ---------------- weight pack: f32 -> zero-padded bf16 ----------------
__global__ __launch_bounds__(256)
void k_pack(const float* __restrict__ W, const float* __restrict__ b,
            __bf16* __restrict__ Wp, float* __restrict__ bp,
            int dout, int din, int doutp, int dinp) {
  int idx = blockIdx.x * 256 + threadIdx.x;
  int total = doutp * dinp;
  if (idx < total) {
    int r = idx / dinp, c = idx - r * dinp;
    float v = (r < dout && c < din) ? W[r * din + c] : 0.f;
    Wp[idx] = (__bf16)v;
  }
  if (idx < doutp) bp[idx] = (idx < dout) ? b[idx] : 0.f;
}

// ---------------- WMMA GEMM: C[M,Npad] = A[M,Kpad] * Wp^T + bias ----------------
// Wp is (Npad x Kpad) row-major bf16 (so B[k][n] = Wp[n][k]).
// Block: 8 waves; wave computes 16(M) x 64(N) via 4 wmma tiles sharing one A fragment.
// The block's 64 x Kpad bf16 B-panel is staged once into LDS by the Tensor Data Mover
// (one descriptor issued by wave 0, tracked with TENSORcnt), then all waves read
// B fragments from LDS instead of 8x-redundant L2 fetches.
__global__ __launch_bounds__(256)
void k_gemm(const float* __restrict__ A,
            const __bf16* __restrict__ W, int ldw,
            const float* __restrict__ bias,
            float* __restrict__ C, int M, int Kpad) {
  extern __shared__ __bf16 smemB[];          // 64 * Kpad bf16 (<= 64 KB)

  int lane = threadIdx.x & 31;
  int wave = threadIdx.x >> 5;
  int half = lane >> 4;        // lane group (K-half selector per ISA 16-bit A/B layout)
  int l16  = lane & 15;
  int mBase = blockIdx.x * 128 + wave * 16;
  int nBase = blockIdx.y * 64;

#if USE_TDM
  if (threadIdx.x < 32) {
    // ---- Tensor DMA descriptor (ISA cdna5 ch.8): 2D tile, 64 rows x Kpad bf16 ----
    unsigned long long gaddr =
        (unsigned long long)(const void*)(W + (size_t)nBase * ldw);
    v4u g0 = {0u, 0u, 0u, 0u};
    g0[0] = 1u;                                   // count=1, user descriptor
    g0[1] = 0u;                                   // lds_addr: dynamic-LDS base (offset 0)
    g0[2] = (unsigned)(gaddr & 0xffffffffu);      // global_addr[31:0]
    g0[3] = (unsigned)((gaddr >> 32) & 0x1ffffffu) | (2u << 30);  // addr[56:32] | type=2

    unsigned kp = (unsigned)Kpad;
    v8i g1 = {0, 0, 0, 0, 0, 0, 0, 0};
    g1[0] = (int)(1u << 16);                      // wg_mask=0, data_size=1 (2 bytes)
    g1[1] = (int)((kp & 0xffffu) << 16);          // tensor_dim0[15:0]   (bits 79:48)
    g1[2] = (int)((kp >> 16) | (64u << 16));      // tensor_dim0[31:16] | tensor_dim1[15:0]
    g1[3] = (int)((kp & 0xffffu) << 16);          // tensor_dim1[31:16]=0 | tile_dim0=Kpad
    g1[4] = (int)64u;                             // tile_dim1=64 rows, tile_dim2=0
    g1[5] = (int)kp;                              // tensor_dim0_stride[31:0] = Kpad
    g1[6] = 0;                                    // stride0[47:32] | stride1[15:0]
    g1[7] = 0;                                    // stride1[47:16]

    v4i gz4 = {0, 0, 0, 0};                       // 2D tensor: groups 2/3 unused
    v8i gz8 = {0, 0, 0, 0, 0, 0, 0, 0};
    __builtin_amdgcn_tensor_load_to_lds(g0, g1, gz4, gz4, gz8, 0);
    __builtin_amdgcn_s_wait_tensorcnt(0);         // data in LDS before barrier signal
  }
  __syncthreads();
#else
  // Fallback: cooperative staging through VGPRs (uint4 = 8 bf16 per thread-chunk)
  {
    int chunks = (64 * Kpad) >> 3;
    const uint4* gsrc = (const uint4*)(W + (size_t)nBase * ldw);
    uint4* ldst = (uint4*)smemB;
    for (int i = threadIdx.x; i < chunks; i += 256) ldst[i] = gsrc[i];
  }
  __syncthreads();
#endif

  int mRow = mBase + l16;
  if (mRow >= M) mRow = M - 1;               // clamp; stores are guarded
  const float* arow = A + (size_t)mRow * STRIDE;

  v8f acc[4] = {};

  for (int k0 = 0; k0 < Kpad; k0 += 32) {
    // A fragment (16x32 bf16): lane half h holds K = {k0+16h..k0+16h+15} (contiguous)
    v8f f0 = *(const v8f*)(arow + k0 + half * 16);
    v8f f1 = *(const v8f*)(arow + k0 + half * 16 + 8);
    __builtin_prefetch(arow + k0 + 32, 0, 3);     // next K-slice (speculative, in-buffer)
    v16bf a;
    #pragma unroll
    for (int i = 0; i < 8; ++i) { a[i] = (__bf16)f0[i]; a[8 + i] = (__bf16)f1[i]; }
    #pragma unroll
    for (int nt = 0; nt < 4; ++nt) {
      // B fragment (32x16 bf16) from LDS: lane = col n, half selects contiguous K 16-chunk
      const __bf16* brow = smemB + (size_t)(nt * 16 + l16) * Kpad + k0 + half * 16;
      v16bf b = *(const v16bf*)brow;
      acc[nt] = __builtin_amdgcn_wmma_f32_16x16x32_bf16(
          false, a, false, b, (short)0, acc[nt], false, false);
    }
  }
  // C/D layout: VGPR r -> row (mBase + 8*half + r), col = nBase + nt*16 + l16
  #pragma unroll
  for (int nt = 0; nt < 4; ++nt) {
    int col = nBase + nt * 16 + l16;
    float bv = bias[col];
    #pragma unroll
    for (int r = 0; r < 8; ++r) {
      int row = mBase + 8 * half + r;
      if (row < M) C[(size_t)row * STRIDE + col] = acc[nt][r] + bv;
    }
  }
}

// ---------------- readout ----------------
__global__ void k_out_init(float* out, const float* cls_b) {
  int i = threadIdx.x;
  if (i < N_GRAPHS) out[i] = cls_b[0];
}

__global__ __launch_bounds__(256)
void k_pool(const float* __restrict__ vt, const int* __restrict__ batch,
            const float* __restrict__ clsW, float* __restrict__ out) {
  int node = blockIdx.x * 8 + (threadIdx.x >> 5);
  if (node >= N_NODES) return;
  int lane = threadIdx.x & 31;
  const float* p = vt + (size_t)node * STRIDE;
  float dot = 0.f;
  for (int i = lane; i < 511; i += 32) dot += p[i] * clsW[1 + i];  // time slot is 0
  dot = wredsum(dot);
  if (lane == 0) atomicAdd(&out[batch[node]], dot);
}

// ---------------- host ----------------
extern "C" void kernel_launch(void* const* d_in, const int* in_sizes, int n_in,
                              void* d_out, int out_size, void* d_ws, size_t ws_size,
                              hipStream_t stream) {
  (void)in_sizes; (void)n_in; (void)out_size; (void)ws_size;
  const float* x     = (const float*)d_in[0];
  const int*   ei    = (const int*)d_in[1];    // [2, N_EDGES]: src then dst
  const int*   batch = (const int*)d_in[2];
  const float* Win[6] = { (const float*)d_in[3],  (const float*)d_in[5],
                          (const float*)d_in[7],  (const float*)d_in[9],
                          (const float*)d_in[11], (const float*)d_in[13] };
  const float* bin[6] = { (const float*)d_in[4],  (const float*)d_in[6],
                          (const float*)d_in[8],  (const float*)d_in[10],
                          (const float*)d_in[12], (const float*)d_in[14] };
  const float* clsW = (const float*)d_in[15];
  const float* clsb = (const float*)d_in[16];
  float* out = (float*)d_out;

  // workspace layout
  float*  bufV  = (float*)d_ws;                               // GEMM inputs / vt
  float*  bufB  = bufV + (size_t)N_NODES * STRIDE;            // agg / GEMM outputs
  __bf16* wbase = (__bf16*)(bufB + (size_t)N_NODES * STRIDE);

  const int dout_ [6] = {127, 127, 255, 255, 511, 511};
  const int din_  [6] = {128, 127, 127, 255, 255, 511};
  const int doutp_[6] = {128, 128, 256, 256, 512, 512};
  const int dinp_ [6] = {128, 128, 128, 256, 256, 512};
  __bf16* Wp[6]; float* bp[6];
  size_t woff = 0;
  for (int i = 0; i < 6; ++i) { Wp[i] = wbase + woff; woff += (size_t)doutp_[i] * dinp_[i]; }
  float* bbase = (float*)(wbase + woff);
  size_t boff = 0;
  for (int i = 0; i < 6; ++i) { bp[i] = bbase + boff; boff += doutp_[i]; }

  // 1) pack weights (tiny)
  for (int i = 0; i < 6; ++i) {
    int total = doutp_[i] * dinp_[i];
    k_pack<<<(total + 255) / 256, 256, 0, stream>>>(
        Win[i], bin[i], Wp[i], bp[i], dout_[i], din_[i], doutp_[i], dinp_[i]);
  }

  const int nodeBlocks = (N_NODES + 7) / 8;

  // 2) embed: vt0 = log_1(exp_1(x)); also init segment_sum buffer with self term
  k_chain<<<nodeBlocks, 256, 0, stream>>>(x, bufV, bufB, IN_FEAT, 128,
                                          1.f, 1.f, 0, 0.f, 0.f);

  const int agglc[3] = {5, 5, 6};          // dpad/4 = 32,32,64 chunks per edge
  const int aggpad[3] = {128, 128, 256};

  for (int L = 0; L < 3; ++L) {
    int w1 = 2 * L, w2 = 2 * L + 1;
    // segment_sum over edges
    int tot = N_EDGES << agglc[L];
    k_scatter<<<(tot + 255) / 256, 256, 0, stream>>>(bufV, bufB, ei, ei + N_EDGES, agglc[L]);
    // u = log_4(exp_1(agg))  -> GEMM1 input
    k_chain<<<nodeBlocks, 256, 0, stream>>>(bufB, bufV, nullptr, STRIDE, aggpad[L],
                                            1.f, 4.f, 0, 0.f, 0.f);
    // GEMM1 (LDS bytes = 64 rows * Kpad * 2)
    dim3 g1((N_NODES + 127) / 128, doutp_[w1] / 64);
    k_gemm<<<g1, 256, (size_t)64 * dinp_[w1] * 2, stream>>>(
        bufV, Wp[w1], dinp_[w1], bp[w1], bufB, N_NODES, dinp_[w1]);
    // mid-MLP: log_4(exp_4(relu(log_4(exp_4(m)))))
    k_chain<<<nodeBlocks, 256, 0, stream>>>(bufB, bufV, nullptr, STRIDE, doutp_[w1],
                                            4.f, 4.f, 1, 4.f, 4.f);
    // GEMM2
    dim3 g2((N_NODES + 127) / 128, doutp_[w2] / 64);
    k_gemm<<<g2, 256, (size_t)64 * dinp_[w2] * 2, stream>>>(
        bufV, Wp[w2], dinp_[w2], bp[w2], bufB, N_NODES, dinp_[w2]);
    // tail: log_1(exp_1(relu(log_4(exp_4(m))))) -> vt_{L+1}; seed next agg if any
    float* ac = (L < 2) ? bufB : nullptr;  // safe: each row fully read before written
    k_chain<<<nodeBlocks, 256, 0, stream>>>(bufB, bufV, ac, STRIDE, doutp_[w2],
                                            4.f, 4.f, 1, 1.f, 1.f);
  }

  // 3) graph readout: out[g] = cls_b + sum_{batch[n]==g} <vt3[n], cls_W[1:]>
  k_out_init<<<1, 64, 0, stream>>>(out, clsb);
  k_pool<<<nodeBlocks, 256, 0, stream>>>(bufV, batch, clsW, out);
}
